// MSDeformAttnTransformerEncoderOnly_51230369906769
// MI455X (gfx1250) — matrix operand order repaired and checked by reading, not compiled, not run on hardware
//
#include <hip/hip_runtime.h>
#include <hip/hip_bf16.h>
#include <hip/hip_fp16.h>

// ---------------- problem constants ----------------
#define D_MODEL 256
#define NHEAD   8
#define NLVL    4
#define NPT     4
#define DHEAD   32
#define NLAYERS 6
#define DFF     1024
#define BATCH   2
#define S_TOT   21760
#define MTOT    (BATCH * S_TOT)          // 43520 rows, divisible by 128

typedef __attribute__((ext_vector_type(16))) _Float16 v16h;
typedef __attribute__((ext_vector_type(8)))  float    v8f;

union Frag16 { v16h v; uint4 u[2]; };

// ---- CDNA5 async global->LDS DMA (ASYNCcnt-tracked), GVS addressing ----
__device__ __forceinline__ void async_ld_b128(const void* gbase, unsigned goff_bytes, void* lds_ptr) {
    unsigned lds_off = (unsigned)(size_t)lds_ptr;   // low 32 bits of flat LDS addr = DS offset
    asm volatile("global_load_async_to_lds_b128 %0, %1, %2"
                 :: "v"(lds_off), "v"(goff_bytes), "s"(gbase)
                 : "memory");
}
__device__ __forceinline__ void wait_asynccnt0() {
    asm volatile("s_wait_asynccnt 0x0" ::: "memory");
}

__device__ __forceinline__ void level_of(int s, int& lvl, int& base, int& H, int& W) {
    if (s < 16384)      { lvl = 0; base = 0;     H = 128; W = 128; }
    else if (s < 20480) { lvl = 1; base = 16384; H = 64;  W = 64;  }
    else if (s < 21504) { lvl = 2; base = 20480; H = 32;  W = 32;  }
    else                { lvl = 3; base = 21504; H = 16;  W = 16;  }
}

// ---------------- weight convert + transpose: W[K,N] f32 -> Wt[N,K] f16 ----------------
__global__ void wcvt_kernel(const float* __restrict__ W, _Float16* __restrict__ Wo, int K, int N) {
    int i = blockIdx.x * blockDim.x + threadIdx.x;
    if (i >= K * N) return;
    int n = i / K, k = i % K;                 // output is N-major
    Wo[i] = (_Float16)W[(size_t)k * N + n];
}

// ---------------- flatten srcs/poss -> [B,S,D] ----------------
__global__ void flatten_kernel(const float* __restrict__ s0, const float* __restrict__ s1,
                               const float* __restrict__ s2, const float* __restrict__ s3,
                               const float* __restrict__ p0, const float* __restrict__ p1,
                               const float* __restrict__ p2, const float* __restrict__ p3,
                               const float* __restrict__ lvl_emb,
                               float* __restrict__ out, float* __restrict__ pos) {
    int i = blockIdx.x * blockDim.x + threadIdx.x;   // exactly MTOT*D_MODEL threads
    int d  = i & (D_MODEL - 1);
    int bs = i >> 8;
    int s  = bs % S_TOT;
    int b  = bs / S_TOT;
    int lvl, base, H, W;
    level_of(s, lvl, base, H, W);
    const float* sp = (lvl == 0) ? s0 : (lvl == 1) ? s1 : (lvl == 2) ? s2 : s3;
    const float* pp = (lvl == 0) ? p0 : (lvl == 1) ? p1 : (lvl == 2) ? p2 : p3;
    int hw  = s - base;
    size_t src_idx = ((size_t)(b * D_MODEL + d) * H + (hw / W)) * W + (hw % W);
    out[i] = sp[src_idx];
    pos[i] = pp[src_idx] + lvl_emb[lvl * D_MODEL + d];
}

// ---------------- WMMA GEMM: C[M,N] = A[M,K] @ Wt^T + bias  (Wt is f16, N-major [N,K]) ----
// AMODE 0: A = f32    | AMODE 1: A = f32 + A2 (q = out + pos) | AMODE 2: A = f16 (async-staged)
// EMODE 0: store f32  | EMODE 1: store f16 with ReLU (FFN hidden)
// block tile 128(M) x 64(N), K staged 32/iter; 8 waves; wave tile 16x64 (4 WMMA / K-step)
template<int AMODE, int EMODE>
__global__ __launch_bounds__(256) void wmma_gemm(const void* __restrict__ Aptr,
                                                 const float* __restrict__ A2,
                                                 const _Float16* __restrict__ Wt,
                                                 const float* __restrict__ bias,
                                                 void* __restrict__ Cptr,
                                                 int M, int N, int K) {
    __shared__ __align__(16) _Float16 sA [128 * 32];   // A tile, row-major [128][32]
    __shared__ __align__(16) _Float16 sBt[ 64 * 32];   // B tile, N-major  [64][32]

    const int tid  = threadIdx.x;
    const int lane = tid & 31;
    const int wv   = tid >> 5;
    const int m0   = blockIdx.x * 128;
    const int n0   = blockIdx.y * 64;

    v8f acc[4] = {v8f{}, v8f{}, v8f{}, v8f{}};

    for (int k0 = 0; k0 < K; k0 += 32) {
        __syncthreads();   // previous iteration's readers done before restaging

        // ---- stage B tile via async DMA: thread t copies 16B (halves [t*8, t*8+8)) ----
        {
            int n  = tid >> 2;                 // 0..63
            int kk = (tid & 3) * 8;            // 0,8,16,24
            unsigned goff = (unsigned)(((size_t)(n0 + n) * K + k0 + kk) * sizeof(_Float16));
            async_ld_b128(Wt, goff, &sBt[tid * 8]);
        }
        // ---- stage A tile ----
        if constexpr (AMODE == 2) {
            // f16 activations: linear copy, async DMA (2 x 16B per thread)
            const _Float16* Ah = (const _Float16*)Aptr;
            int r  = tid >> 1;                 // 0..127
            int kk = (tid & 1) * 16;           // 0,16
            unsigned g0 = (unsigned)(((size_t)(m0 + r) * K + k0 + kk) * sizeof(_Float16));
            async_ld_b128(Ah, g0,      &sA[tid * 16]);
            async_ld_b128(Ah, g0 + 16, &sA[tid * 16 + 8]);
        } else {
            for (int i = tid; i < 128 * 32; i += 256) {
                int r = i >> 5, c = i & 31;
                size_t gi = (size_t)(m0 + r) * K + k0 + c;
                float v = ((const float*)Aptr)[gi];
                if constexpr (AMODE == 1) v += A2[gi];
                sA[r * 32 + c] = (_Float16)v;
            }
        }
        wait_asynccnt0();      // this wave's DMA done
        __syncthreads();       // all waves' LDS writes visible

        // A fragment: 16-bit 16x32 layout -> lanes 0-15 K{0..7,16..23}, lanes 16-31 K{8..15,24..31}
        int r    = (wv << 4) + (lane & 15);
        int ksel = (lane & 16) ? 8 : 0;
        Frag16 fa;
        fa.u[0] = *reinterpret_cast<const uint4*>(&sA[r * 32 + ksel]);
        fa.u[1] = *reinterpret_cast<const uint4*>(&sA[r * 32 + 16 + ksel]);
        // B fragments: lanes 0-15 hold K{0..7,16..23} of column n; lanes 16-31 hold K{8..15,24..31}
        int nrow = lane & 15;
        int kb   = (lane & 16) ? 8 : 0;
#pragma unroll
        for (int j = 0; j < 4; ++j) {
            Frag16 fb;
            fb.u[0] = *reinterpret_cast<const uint4*>(&sBt[(16 * j + nrow) * 32 + kb]);
            fb.u[1] = *reinterpret_cast<const uint4*>(&sBt[(16 * j + nrow) * 32 + kb + 16]);
            acc[j] = __builtin_amdgcn_wmma_f32_16x16x32_f16(false, fa.v, false, fb.v,
                                                            (short)0, acc[j], false, false);
        }
    }

    // C/D layout: VGPR v -> M = v (+8 for lanes 16-31), N = lane & 15
    int rowbase = m0 + (wv << 4) + ((lane & 16) ? 8 : 0);
    int cl = lane & 15;
    float bv[4];
#pragma unroll
    for (int j = 0; j < 4; ++j) bv[j] = bias[n0 + 16 * j + cl];
#pragma unroll
    for (int v = 0; v < 8; ++v) {
        int row = rowbase + v;
#pragma unroll
        for (int j = 0; j < 4; ++j) {
            size_t ix = (size_t)row * N + n0 + 16 * j + cl;
            float o = acc[j][v] + bv[j];
            if constexpr (EMODE == 0) ((float*)Cptr)[ix] = o;
            else                      ((_Float16*)Cptr)[ix] = (_Float16)fmaxf(o, 0.f);
        }
    }
}

// ---------------- softmax over 16 (level,point) logits per (token,head) ----------------
__global__ void softmax16_kernel(float* __restrict__ attn, int total) {
    int t = blockIdx.x * blockDim.x + threadIdx.x;
    if (t >= total) return;
    int bs = t / NHEAD, h = t % NHEAD;
    float* p = attn + (size_t)bs * (NHEAD * NLVL * NPT) + h * 16;
    float v[16], m = -1e30f;
    for (int j = 0; j < 16; ++j) { v[j] = p[j]; m = fmaxf(m, v[j]); }
    float s = 0.f;
    for (int j = 0; j < 16; ++j) { v[j] = __expf(v[j] - m); s += v[j]; }
    float inv = 1.f / s;
    for (int j = 0; j < 16; ++j) p[j] = v[j] * inv;
}

// ---------------- deformable bilinear sampling: one wave per (b,s,head), lane = channel ----------------
__global__ void deform_sample_kernel(const float* __restrict__ value,
                                     const float* __restrict__ off,
                                     const float* __restrict__ attn,
                                     float* __restrict__ sampled) {
    int wave = blockIdx.x * 8 + (threadIdx.x >> 5);
    int lane = threadIdx.x & 31;              // channel within head (DHEAD == 32)
    int h  = wave & (NHEAD - 1);
    int bs = wave >> 3;                        // b*S + s
    int s  = bs % S_TOT;
    int b  = bs / S_TOT;

    int lq, baseq, Hq, Wq;
    level_of(s, lq, baseq, Hq, Wq);
    int hw = s - baseq;
    float ref_x = ((hw % Wq) + 0.5f) / (float)Wq;
    float ref_y = ((hw / Wq) + 0.5f) / (float)Hq;

    const float* offp = off  + (size_t)bs * 256 + h * 32;  // [NH,NL,NP,2] within token
    const float* awp  = attn + (size_t)bs * 128 + h * 16;  // [NH,NL*NP] within token

    const int LB[4] = {0, 16384, 20480, 21504};
    const int LS[4] = {128, 64, 32, 16};

    float acc = 0.f;
    for (int l = 0; l < NLVL; ++l) {
        int Hl = LS[l], Wl = LS[l];
        const float* vbase = value + ((size_t)(b * S_TOT + LB[l]) * NHEAD + h) * DHEAD + lane;
        for (int p = 0; p < NPT; ++p) {
            float ox = offp[l * 8 + p * 2 + 0];
            float oy = offp[l * 8 + p * 2 + 1];
            float aw = awp[l * 4 + p];
            float x = (ref_x + ox / (float)Wl) * Wl - 0.5f;
            float y = (ref_y + oy / (float)Hl) * Hl - 0.5f;
            float xf = floorf(x), yf = floorf(y);
            float lx = x - xf, ly = y - yf;
            int x0 = (int)xf, y0 = (int)yf;
            float w00 = (1.f - lx) * (1.f - ly);
            float w10 = lx * (1.f - ly);
            float w01 = (1.f - lx) * ly;
            float w11 = lx * ly;
            float samp = 0.f;
            if (x0     >= 0 && x0     < Wl && y0     >= 0 && y0     < Hl)
                samp += w00 * vbase[(size_t)(y0 * Wl + x0) * (NHEAD * DHEAD)];
            if (x0 + 1 >= 0 && x0 + 1 < Wl && y0     >= 0 && y0     < Hl)
                samp += w10 * vbase[(size_t)(y0 * Wl + x0 + 1) * (NHEAD * DHEAD)];
            if (x0     >= 0 && x0     < Wl && y0 + 1 >= 0 && y0 + 1 < Hl)
                samp += w01 * vbase[(size_t)((y0 + 1) * Wl + x0) * (NHEAD * DHEAD)];
            if (x0 + 1 >= 0 && x0 + 1 < Wl && y0 + 1 >= 0 && y0 + 1 < Hl)
                samp += w11 * vbase[(size_t)((y0 + 1) * Wl + x0 + 1) * (NHEAD * DHEAD)];
            acc += aw * samp;
        }
    }
    sampled[(size_t)bs * D_MODEL + h * DHEAD + lane] = acc;
}

// ---------------- residual add + LayerNorm: one wave per token row ----------------
__global__ void add_ln_kernel(const float* __restrict__ resid,
                              const float* __restrict__ delta,
                              const float* __restrict__ gamma,
                              const float* __restrict__ beta,
                              float* __restrict__ dst) {
    int wave = blockIdx.x * 8 + (threadIdx.x >> 5);
    int lane = threadIdx.x & 31;
    size_t base = (size_t)wave * D_MODEL;
    float x[8];
    float s = 0.f, sq = 0.f;
    for (int j = 0; j < 8; ++j) {
        int c = lane + j * 32;
        x[j] = resid[base + c] + delta[base + c];
        s  += x[j];
        sq += x[j] * x[j];
    }
    for (int o = 16; o > 0; o >>= 1) {
        s  += __shfl_xor(s,  o, 32);
        sq += __shfl_xor(sq, o, 32);
    }
    float mean = s * (1.f / D_MODEL);
    float var  = sq * (1.f / D_MODEL) - mean * mean;
    float inv  = rsqrtf(var + 1e-5f);
    for (int j = 0; j < 8; ++j) {
        int c = lane + j * 32;
        dst[base + c] = (x[j] - mean) * inv * gamma[c] + beta[c];
    }
}

// ---------------- host orchestration ----------------
extern "C" void kernel_launch(void* const* d_in, const int* in_sizes, int n_in,
                              void* d_out, int out_size, void* d_ws, size_t ws_size,
                              hipStream_t stream) {
    (void)in_sizes; (void)n_in; (void)out_size; (void)ws_size;

    const float* src0 = (const float*)d_in[0];
    const float* src1 = (const float*)d_in[1];
    const float* src2 = (const float*)d_in[2];
    const float* src3 = (const float*)d_in[3];
    const float* pos0 = (const float*)d_in[4];
    const float* pos1 = (const float*)d_in[5];
    const float* pos2 = (const float*)d_in[6];
    const float* pos3 = (const float*)d_in[7];
    const float* lvl_emb = (const float*)d_in[8];
    const float* Woff  = (const float*)d_in[9];
    const float* boff  = (const float*)d_in[10];
    const float* Wattn = (const float*)d_in[11];
    const float* battn = (const float*)d_in[12];
    const float* Wval  = (const float*)d_in[13];
    const float* bval  = (const float*)d_in[14];
    const float* Wout  = (const float*)d_in[15];
    const float* bout  = (const float*)d_in[16];
    const float* ln1s  = (const float*)d_in[17];
    const float* ln1b  = (const float*)d_in[18];
    const float* W1    = (const float*)d_in[19];
    const float* b1    = (const float*)d_in[20];
    const float* W2    = (const float*)d_in[21];
    const float* b2    = (const float*)d_in[22];
    const float* ln2s  = (const float*)d_in[23];
    const float* ln2b  = (const float*)d_in[24];

    // workspace carve-out (~388 MB total)
    char* ws = (char*)d_ws;
    const size_t szBSD = (size_t)MTOT * D_MODEL * sizeof(float);   // 44.56 MB
    float* out     = (float*)ws;  ws += szBSD;
    float* pos     = (float*)ws;  ws += szBSD;
    float* value   = (float*)ws;  ws += szBSD;
    float* offb    = (float*)ws;  ws += szBSD;                      // [B,S,256]
    float* attnb   = (float*)ws;  ws += (size_t)MTOT * 128 * sizeof(float);
    float* sampled = (float*)ws;  ws += szBSD;
    float* tmp     = (float*)ws;  ws += szBSD;
    _Float16* hbuf = (_Float16*)ws; ws += (size_t)MTOT * DFF * sizeof(_Float16);
    _Float16* wbuf = (_Float16*)ws; ws += (size_t)NLAYERS * 753664 * sizeof(_Float16);

    // per-layer f16 transposed-weight offsets (halves)
    const size_t oWval = 0, oWoff = 65536, oWattn = 131072, oWout = 163840,
                 oW1 = 229376, oW2 = 491520, LSTRIDE = 753664;

    // pre-convert + pre-transpose all weights to f16 [N,K] (once per launch)
    for (int i = 0; i < NLAYERS; ++i) {
        _Float16* lw = wbuf + (size_t)i * LSTRIDE;
        wcvt_kernel<<<(65536 + 255) / 256, 256, 0, stream>>>(Wval  + (size_t)i * 65536,  lw + oWval,  256, 256);
        wcvt_kernel<<<(65536 + 255) / 256, 256, 0, stream>>>(Woff  + (size_t)i * 65536,  lw + oWoff,  256, 256);
        wcvt_kernel<<<(32768 + 255) / 256, 256, 0, stream>>>(Wattn + (size_t)i * 32768,  lw + oWattn, 256, 128);
        wcvt_kernel<<<(65536 + 255) / 256, 256, 0, stream>>>(Wout  + (size_t)i * 65536,  lw + oWout,  256, 256);
        wcvt_kernel<<<(262144 + 255) / 256, 256, 0, stream>>>(W1   + (size_t)i * 262144, lw + oW1,    256, 1024);
        wcvt_kernel<<<(262144 + 255) / 256, 256, 0, stream>>>(W2   + (size_t)i * 262144, lw + oW2,    1024, 256);
    }

    flatten_kernel<<<MTOT, 256, 0, stream>>>(src0, src1, src2, src3,
                                             pos0, pos1, pos2, pos3,
                                             lvl_emb, out, pos);

    const dim3 blk(256);
    const dim3 gD(MTOT / 128, D_MODEL / 64);   // N = 256
    const dim3 gA(MTOT / 128, 128 / 64);       // N = 128
    const dim3 gF(MTOT / 128, DFF / 64);       // N = 1024

    for (int i = 0; i < NLAYERS; ++i) {
        const _Float16* lw = wbuf + (size_t)i * LSTRIDE;

        // value = out @ Wval + bval
        wmma_gemm<0, 0><<<gD, blk, 0, stream>>>(out, nullptr, lw + oWval,
                                                bval + (size_t)i * 256, value, MTOT, 256, 256);
        // off = (out + pos) @ Woff + boff
        wmma_gemm<1, 0><<<gD, blk, 0, stream>>>(out, pos, lw + oWoff,
                                                boff + (size_t)i * 256, offb, MTOT, 256, 256);
        // attn logits = (out + pos) @ Wattn + battn
        wmma_gemm<1, 0><<<gA, blk, 0, stream>>>(out, pos, lw + oWattn,
                                                battn + (size_t)i * 128, attnb, MTOT, 128, 256);
        // softmax over 16 per (token, head)
        softmax16_kernel<<<(MTOT * NHEAD + 255) / 256, 256, 0, stream>>>(attnb, MTOT * NHEAD);
        // deformable bilinear gather
        deform_sample_kernel<<<MTOT * NHEAD / 8, 256, 0, stream>>>(value, offb, attnb, sampled);
        // attn = sampled @ Wout + bout
        wmma_gemm<0, 0><<<gD, blk, 0, stream>>>(sampled, nullptr, lw + oWout,
                                                bout + (size_t)i * 256, tmp, MTOT, 256, 256);
        // out = LN1(out + attn)
        add_ln_kernel<<<MTOT / 8, 256, 0, stream>>>(out, tmp, ln1s + i * D_MODEL, ln1b + i * D_MODEL, out);
        // h = relu(out @ W1 + b1) stored as f16
        wmma_gemm<0, 1><<<gF, blk, 0, stream>>>(out, nullptr, lw + oW1,
                                                b1 + (size_t)i * DFF, hbuf, MTOT, DFF, 256);
        // ff = h @ W2 + b2  (f16 A staged via async DMA)
        wmma_gemm<2, 0><<<gD, blk, 0, stream>>>(hbuf, nullptr, lw + oW2,
                                                b2 + (size_t)i * 256, tmp, MTOT, 256, DFF);
        // out = LN2(out + ff)  (last layer writes straight to d_out)
        float* dst = (i == NLAYERS - 1) ? (float*)d_out : out;
        add_ln_kernel<<<MTOT / 8, 256, 0, stream>>>(out, tmp, ln2s + i * D_MODEL, ln2b + i * D_MODEL, dst);
    }
}
